// Translator_76562087019185
// MI455X (gfx1250) — matrix operand order repaired and checked by reading, not compiled, not run on hardware
//
#include <hip/hip_runtime.h>
#include <hip/hip_bf16.h>

// ---------------------------------------------------------------------------
// Pointer-generator beam-search head for MI455X (gfx1250, wave32, WMMA).
// Only t = T-1 contributes to outputs, so we compute an 8x50257 logits slab
// via bf16 WMMA (M padded to 16), then softmax / scatter / top-k.
// w_proj (206 MB) is the only large operand: streamed once, non-temporal.
// ---------------------------------------------------------------------------

typedef __attribute__((ext_vector_type(16))) __bf16 v16bf;
typedef __attribute__((ext_vector_type(8)))  float  v8f;

#define B_    8
#define T_    64
#define D_    1024
#define V_    50257
#define H_    8
#define S_    400
#define NOOV  50
#define VTOT  (V_ + NOOV)          // 50307
#define VP    50304                // padded logits row stride
#define FP    50432                // padded final-dist row stride (mult of 256)
#define BEAM  8

// workspace layout (float offsets)
#define OFF_LOGITS 0
#define OFF_FINAL  (8 * VP)                  // 402432
#define OFF_STATS  (OFF_FINAL + 8 * FP)      // [b*4+0]=rowmax [1]=rowsum [2]=p_gen
#define OFF_ATTN   (OFF_STATS + 32)          // 8 x 512 (attn_score * (1-p_gen))
#define OFF_TOPKP  (OFF_ATTN + 8 * 512)      // 8 x 8 probs
#define OFF_TOPKI  (OFF_TOPKP + 64)          // 8 x 8 indices (as float)

static __device__ __forceinline__ __bf16 f2bf(float f) {
    unsigned u = __builtin_bit_cast(unsigned, f);
    unsigned r = (u + 0x7fffu + ((u >> 16) & 1u)) >> 16;   // round-to-nearest-even
    unsigned short h = (unsigned short)r;
    return __builtin_bit_cast(__bf16, h);
}

// ---------------------------------------------------------------------------
// Kernel 1: logits[8][V] = dec_output[:,63,:] (8x1024, padded to 16) @ w_proj.
// One wave owns one 16-column tile; 32 WMMA 16x16x32 bf16 steps over K=1024,
// unrolled x2 so the scheduler overlaps step k+1 loads with step k WMMA.
// ---------------------------------------------------------------------------
__global__ void k1_gemv_wmma(const float* __restrict__ dec,
                             const float* __restrict__ w,
                             float* __restrict__ ws)
{
    const int wave   = threadIdx.x >> 5;
    const int lane   = threadIdx.x & 31;
    const int tile   = blockIdx.x * (blockDim.x >> 5) + wave;
    const int ntiles = (V_ + 15) / 16;
    if (tile >= ntiles) return;          // wave-uniform: EXEC stays all-ones per wave

    const int g  = lane >> 4;            // lane half (0/1)
    const int ml = lane & 15;            // A-row (batch) for lanes; B column within tile
    const int v0 = tile * 16;
    const int v  = v0 + ml;
    const int vc = (v < V_) ? v : (V_ - 1);

    const float* decrow = dec + (size_t)(T_ - 1) * D_;   // + ml*T_*D_ per lane below

    v8f c = {};
    #pragma unroll 2
    for (int kb = 0; kb < D_; kb += 32) {
        if (kb + 32 < D_)
            __builtin_prefetch(&w[(size_t)(kb + 32) * V_ + vc], 0, 0); // global_prefetch_b8

        v16bf a, b;
        #pragma unroll
        for (int e = 0; e < 16; ++e) {
            // A (16x32 bf16): lanes 0-15 hold K=g*8+e (V0-3) and K=16+g*8+(e-8) (V4-7)
            const int ka = kb + ((e >> 3) << 4) + g * 8 + (e & 7);
            const float af = (ml < B_)
                ? decrow[(size_t)ml * T_ * D_ + ka]
                : 0.0f;                       // rows 8..15 are zero padding
            a[e] = f2bf(af);
            // B (32x16 bf16): same K order as A: K=g*8+e (e<8), K=16+g*8+(e-8) (e>=8), N=ml
            const int kbk = kb + ((e >> 3) << 4) + g * 8 + (e & 7);
            // single-use 206 MB stream -> non-temporal to spare L2 for hot data
            b[e] = f2bf(__builtin_nontemporal_load(&w[(size_t)kbk * V_ + vc]));
        }
        c = __builtin_amdgcn_wmma_f32_16x16x32_bf16(false, a, false, b,
                                                    (short)0, c, false, false);
    }

    // C layout: VGPR r, lanes 0-15 -> M=r (the 8 real rows), N=lane
    if (g == 0 && v < V_) {
        #pragma unroll
        for (int r = 0; r < B_; ++r)
            ws[OFF_LOGITS + (size_t)r * VP + v] = c[r];
    }
}

// ---------------------------------------------------------------------------
// Kernel 2: per-row softmax stats, p_gen, and attention softmax (t=63).
// ---------------------------------------------------------------------------
__global__ void k2_stats(const float* __restrict__ attn,
                         const float* __restrict__ pgl,
                         float* __restrict__ ws)
{
    const int b = blockIdx.x;
    const int t = threadIdx.x;     // 256 threads
    __shared__ float red[256];
    __shared__ float aval[512];

    const float* lg = ws + OFF_LOGITS + (size_t)b * VP;

    // row max
    float m = -3.4e38f;
    for (int vv = t; vv < V_; vv += 256) m = fmaxf(m, lg[vv]);
    red[t] = m; __syncthreads();
    for (int st = 128; st > 0; st >>= 1) {
        if (t < st) red[t] = fmaxf(red[t], red[t + st]);
        __syncthreads();
    }
    const float rowmax = red[0]; __syncthreads();

    // row sum of exp
    float sum = 0.0f;
    for (int vv = t; vv < V_; vv += 256) sum += __expf(lg[vv] - rowmax);
    red[t] = sum; __syncthreads();
    for (int st = 128; st > 0; st >>= 1) {
        if (t < st) red[t] += red[t + st];
        __syncthreads();
    }
    const float rowsum = red[0]; __syncthreads();

    const float pg = 1.0f / (1.0f + __expf(-pgl[b * T_ + (T_ - 1)]));
    if (t == 0) {
        ws[OFF_STATS + b * 4 + 0] = rowmax;
        ws[OFF_STATS + b * 4 + 1] = rowsum;
        ws[OFF_STATS + b * 4 + 2] = pg;
    }

    // attention: sum over heads at t=63, softmax over S=400
    const int s0 = t, s1 = t + 256;
    float a0 = -3.4e38f, a1 = -3.4e38f;
    if (s0 < S_) {
        float acc = 0.0f;
        for (int h = 0; h < H_; ++h)
            acc += attn[(((size_t)b * H_ + h) * T_ + (T_ - 1)) * S_ + s0];
        aval[s0] = acc; a0 = acc;
    }
    if (s1 < S_) {
        float acc = 0.0f;
        for (int h = 0; h < H_; ++h)
            acc += attn[(((size_t)b * H_ + h) * T_ + (T_ - 1)) * S_ + s1];
        aval[s1] = acc; a1 = acc;
    }
    red[t] = fmaxf(a0, a1); __syncthreads();
    for (int st = 128; st > 0; st >>= 1) {
        if (t < st) red[t] = fmaxf(red[t], red[t + st]);
        __syncthreads();
    }
    const float amax = red[0]; __syncthreads();

    const float e0 = (s0 < S_) ? __expf(aval[s0] - amax) : 0.0f;
    const float e1 = (s1 < S_) ? __expf(aval[s1] - amax) : 0.0f;
    red[t] = e0 + e1; __syncthreads();
    for (int st = 128; st > 0; st >>= 1) {
        if (t < st) red[t] += red[t + st];
        __syncthreads();
    }
    const float adenom = red[0]; __syncthreads();

    const float scale = (1.0f - pg) / adenom;
    if (s0 < S_) ws[OFF_ATTN + b * 512 + s0] = e0 * scale;
    if (s1 < S_) ws[OFF_ATTN + b * 512 + s1] = e1 * scale;
}

// ---------------------------------------------------------------------------
// Kernel 3: final[b][v] = p_gen * softmax(logits)[v] (OOV slots = 0).
// ---------------------------------------------------------------------------
__global__ void k3_final(float* __restrict__ ws)
{
    const int b = blockIdx.y;
    const int v = blockIdx.x * 256 + threadIdx.x;
    const float rowmax = ws[OFF_STATS + b * 4 + 0];
    const float rowsum = ws[OFF_STATS + b * 4 + 1];
    const float pg     = ws[OFF_STATS + b * 4 + 2];
    float out = 0.0f;
    if (v < V_)
        out = __expf(ws[OFF_LOGITS + (size_t)b * VP + v] - rowmax) / rowsum * pg;
    if (v < FP)
        ws[OFF_FINAL + (size_t)b * FP + v] = out;
}

// ---------------------------------------------------------------------------
// Kernel 3b: scatter-add copy distribution at src_idx.
// ---------------------------------------------------------------------------
__global__ void k3b_scatter(const int* __restrict__ src_idx, float* __restrict__ ws)
{
    const int b = blockIdx.x;
    const int s = threadIdx.x;
    if (s < S_) {
        const int idx = src_idx[s];
        atomicAdd(&ws[OFF_FINAL + (size_t)b * FP + idx], ws[OFF_ATTN + b * 512 + s]);
    }
}

// ---------------------------------------------------------------------------
// Kernel 4: top-8 per row by iterative argmax (mask-out with -inf).
// ---------------------------------------------------------------------------
__global__ void k4_topk(float* __restrict__ ws)
{
    const int b = blockIdx.x;
    const int t = threadIdx.x;     // 256
    __shared__ float sv[256];
    __shared__ int   si[256];
    float* fin = ws + OFF_FINAL + (size_t)b * FP;

    for (int k = 0; k < BEAM; ++k) {
        float bv = -3.4e38f; int bi = 0;
        for (int v = t; v < VTOT; v += 256) {
            const float x = fin[v];
            if (x > bv) { bv = x; bi = v; }
        }
        sv[t] = bv; si[t] = bi; __syncthreads();
        for (int st = 128; st > 0; st >>= 1) {
            if (t < st) {
                if (sv[t + st] > sv[t] ||
                    (sv[t + st] == sv[t] && si[t + st] < si[t])) {
                    sv[t] = sv[t + st]; si[t] = si[t + st];
                }
            }
            __syncthreads();
        }
        if (t == 0) {
            ws[OFF_TOPKP + b * 8 + k] = sv[0];
            ws[OFF_TOPKI + b * 8 + k] = (float)si[0];
            fin[si[0]] = -3.4e38f;     // recomputed by k3 every call -> deterministic
        }
        __syncthreads();
    }
}

// ---------------------------------------------------------------------------
// Kernel 5: beam combine: top-8 of log(p)+score over 64 candidates.
// Outputs (flat, floats): new_scores[8] | best_k_idx[8] | best_r[8].
// ---------------------------------------------------------------------------
__global__ void k5_combine(const float* __restrict__ scores,
                           float* __restrict__ out,
                           const float* __restrict__ ws)
{
    if (threadIdx.x != 0) return;
    float cand[64];
    for (int i = 0; i < 64; ++i)
        cand[i] = __logf(ws[OFF_TOPKP + i]) + scores[i / 8];
    bool used[64] = {};
    for (int k = 0; k < BEAM; ++k) {
        int bi = 0; float bv = -3.4e38f;
        for (int i = 0; i < 64; ++i)
            if (!used[i] && cand[i] > bv) { bv = cand[i]; bi = i; }
        used[bi] = true;
        const int r = bi / BEAM, c = bi % BEAM;
        out[k]      = bv;
        out[8 + k]  = ws[OFF_TOPKI + r * 8 + c];
        out[16 + k] = (float)r;
    }
}

// ---------------------------------------------------------------------------
extern "C" void kernel_launch(void* const* d_in, const int* in_sizes, int n_in,
                              void* d_out, int out_size, void* d_ws, size_t ws_size,
                              hipStream_t stream)
{
    const float* dec    = (const float*)d_in[0];  // (8,64,1024)
    const float* w      = (const float*)d_in[1];  // (1024,50257)
    const float* attn   = (const float*)d_in[2];  // (8,8,64,400)
    const float* pgl    = (const float*)d_in[3];  // (8,64,1)
    const int*   sidx   = (const int*)d_in[4];    // (1,400)
    const float* scores = (const float*)d_in[5];  // (8,)
    float* out = (float*)d_out;
    float* ws  = (float*)d_ws;

    const int ntiles = (V_ + 15) / 16;   // 3142 column tiles
    const int wpb    = 8;                // waves per block (256 threads)
    k1_gemv_wmma<<<dim3((ntiles + wpb - 1) / wpb), wpb * 32, 0, stream>>>(dec, w, ws);
    k2_stats<<<dim3(B_), 256, 0, stream>>>(attn, pgl, ws);
    k3_final<<<dim3(FP / 256, B_), 256, 0, stream>>>(ws);
    k3b_scatter<<<dim3(B_), 512, 0, stream>>>(sidx, ws);
    k4_topk<<<dim3(B_), 256, 0, stream>>>(ws);
    k5_combine<<<1, 32, 0, stream>>>(scores, out, ws);
}
